// TemporalGCN2_28724741276162
// MI455X (gfx1250) — matrix operand order repaired and checked by reading, not compiled, not run on hardware
//
#include <hip/hip_runtime.h>
#include <hip/hip_bf16.h>
#include <cstdint>

typedef __attribute__((ext_vector_type(16))) __bf16 v16bf;
typedef __attribute__((ext_vector_type(8)))  __bf16 v8bf;
typedef __attribute__((ext_vector_type(8)))  float  v8f;

#define T_STEPS 24
#define NNODES  20000
#define DIN     64
#define HTOT    128
#define G3H     384
#define NEDGE   640000
#define NEG_SLOPE 0.01f

// ---------------------------------------------------------------- utilities

__device__ __forceinline__ void atomicMaxF(float* a, float v) {
    int* ai = reinterpret_cast<int*>(a);
    int cur = *ai;
    while (__int_as_float(cur) < v) {
        int assumed = cur;
        cur = atomicCAS(ai, assumed, __float_as_int(v));
        if (cur == assumed) break;
    }
}

__global__ void fill_kernel(float* __restrict__ p, float v, int n) {
    int i = blockIdx.x * blockDim.x + threadIdx.x;
    if (i < n) p[i] = v;
}

__global__ void scale_kernel(float* __restrict__ dst, const float* __restrict__ src,
                             float s, int n) {
    int i = blockIdx.x * blockDim.x + threadIdx.x;
    if (i < n) dst[i] = src[i] * s;
}

// ---------------------------------------------------------------- bf16 WMMA GEMM
// C[M,N] = A[M,K] * B + bias   (fp32 in, bf16 WMMA fragments, fp32 accumulate)
// BT==true : B stored [N,K] (GRU weights, x @ W^T). Hot path (48 launches):
//            tile is contiguous -> cooperative load + convert ONCE to bf16 in
//            LDS; B fragment = two ds_load_b128 straight into the WMMA operand.
// BT==false: B stored [K,N] (conv weights). Transpose gather: async
//            global->LDS b32 (no VGPR round trip, ASYNCcnt-tracked), fp32 in
//            LDS, convert at fragment read. Only 4 launches.
// Block = 256 thr = 8 waves; block tile 128 rows x 64 cols; each wave computes
// a 16x64 strip = 4 WMMA per 32-wide K step sharing one A fragment.
template<int K, bool BT>
__global__ __launch_bounds__(256)
void gemm_wmma(const float* __restrict__ A, const float* __restrict__ B,
               const float* __restrict__ bias, float* __restrict__ C,
               int M, int N)
{
    constexpr int SKH = K + 8;                    // bf16 row stride (BT path)
    constexpr int SKF = K + 4;                    // fp32 row stride (!BT path)
    constexpr int BSBYTES = BT ? 64 * SKH * 2 : 64 * SKF * 4;
    __shared__ __align__(16) unsigned char BsRaw[BSBYTES];
    const int tid = threadIdx.x;
    const int n0  = blockIdx.y * 64;

    if constexpr (BT) {
        // convert once per block: float4 pairs -> 8 bf16 -> ds_store_b128
        __bf16* Bh = reinterpret_cast<__bf16*>(BsRaw);
        #pragma unroll
        for (int r = 0; r < K / 32; ++r) {        // 64*(K/8) chunks / 256 threads
            int idx = r * 256 + tid;
            int c  = idx / (K / 8);
            int qd = idx % (K / 8);
            const float* gp = B + (size_t)(n0 + c) * K + qd * 8;
            float4 f0 = *(const float4*)(gp);
            float4 f1 = *(const float4*)(gp + 4);
            v8bf pk;
            pk[0]=(__bf16)f0.x; pk[1]=(__bf16)f0.y; pk[2]=(__bf16)f0.z; pk[3]=(__bf16)f0.w;
            pk[4]=(__bf16)f1.x; pk[5]=(__bf16)f1.y; pk[6]=(__bf16)f1.z; pk[7]=(__bf16)f1.w;
            *reinterpret_cast<v8bf*>(&Bh[c * SKH + qd * 8]) = pk;
        }
    } else {
        // async transpose gather: Bs[c][k] = B[k*N + n0 + c]
        float* Bf = reinterpret_cast<float*>(BsRaw);
        #pragma unroll
        for (int r = 0; r < (64 * K) / 256; ++r) {
            int idx = r * 256 + tid;
            int kq = idx >> 6, c = idx & 63;
            unsigned lds = (unsigned)(size_t)(&Bf[c * SKF + kq]);
            unsigned long long ga =
                (unsigned long long)(size_t)(B + (size_t)kq * N + n0 + c);
            asm volatile("global_load_async_to_lds_b32 %0, %1, off"
                         :: "v"(lds), "v"(ga) : "memory");
        }
        asm volatile("s_wait_asynccnt 0" ::: "memory");
    }
    __syncthreads();

    const int lane = tid & 31;
    const int wave = tid >> 5;
    const int m0   = blockIdx.x * 128 + wave * 16;
    if (m0 >= M) return;                 // wave-uniform: EXEC stays all-ones
    const int lh   = lane >> 4;
    const int l15  = lane & 15;
    const int arow = m0 + l15;

    v8f acc[4] = {};
    #pragma unroll
    for (int kk = 0; kk < K; kk += 32) {
        // A fragment: 16x32 bf16. lanes 0-15: K kk..kk+15,
        //                         lanes16-31: K kk+16..kk+31
        v16bf af;
        {
            const float* ap = A + (size_t)arow * K + kk + lh * 16;
            float4 a0 = *(const float4*)(ap);
            float4 a1 = *(const float4*)(ap + 4);
            float4 a2 = *(const float4*)(ap + 8);
            float4 a3 = *(const float4*)(ap + 12);
            af[0]=(__bf16)a0.x; af[1]=(__bf16)a0.y; af[2]=(__bf16)a0.z; af[3]=(__bf16)a0.w;
            af[4]=(__bf16)a1.x; af[5]=(__bf16)a1.y; af[6]=(__bf16)a1.z; af[7]=(__bf16)a1.w;
            af[8]=(__bf16)a2.x; af[9]=(__bf16)a2.y; af[10]=(__bf16)a2.z; af[11]=(__bf16)a2.w;
            af[12]=(__bf16)a3.x; af[13]=(__bf16)a3.y; af[14]=(__bf16)a3.z; af[15]=(__bf16)a3.w;
        }
        const int kb = kk + lh * 16;
        #pragma unroll
        for (int j = 0; j < 4; ++j) {
            // B fragment: lanes 0-15 hold K=kb..kb+15 for column c=j*16+l15
            const int c = j * 16 + l15;
            v16bf bf;
            if constexpr (BT) {
                const __bf16* bp = reinterpret_cast<const __bf16*>(BsRaw) + c * SKH + kb;
                v8bf blo = *reinterpret_cast<const v8bf*>(bp);
                v8bf bhi = *reinterpret_cast<const v8bf*>(bp + 8);
                bf = __builtin_shufflevector(blo, bhi,
                        0,1,2,3,4,5,6,7,8,9,10,11,12,13,14,15);
            } else {
                const float* bp = reinterpret_cast<const float*>(BsRaw) + c * SKF + kb;
                float4 b0 = *(const float4*)(bp);
                float4 b1 = *(const float4*)(bp + 4);
                float4 b2 = *(const float4*)(bp + 8);
                float4 b3 = *(const float4*)(bp + 12);
                bf[0]=(__bf16)b0.x; bf[1]=(__bf16)b0.y; bf[2]=(__bf16)b0.z; bf[3]=(__bf16)b0.w;
                bf[4]=(__bf16)b1.x; bf[5]=(__bf16)b1.y; bf[6]=(__bf16)b1.z; bf[7]=(__bf16)b1.w;
                bf[8]=(__bf16)b2.x; bf[9]=(__bf16)b2.y; bf[10]=(__bf16)b2.z; bf[11]=(__bf16)b2.w;
                bf[12]=(__bf16)b3.x; bf[13]=(__bf16)b3.y; bf[14]=(__bf16)b3.z; bf[15]=(__bf16)b3.w;
            }
            acc[j] = __builtin_amdgcn_wmma_f32_16x16x32_bf16(
                         false, af, false, bf, (short)0, acc[j], false, false);
        }
    }
    #pragma unroll
    for (int j = 0; j < 4; ++j) {
        const int ncol = n0 + j * 16 + l15;
        const float bv = bias[ncol];
        float* cp = C + (size_t)(m0 + lh * 8) * N + ncol;   // VGPR r -> row m0+lh*8+r
        #pragma unroll
        for (int r = 0; r < 8; ++r) cp[(size_t)r * N] = acc[j][r] + bv;
    }
}

// ---------------------------------------------------------------- GRU elementwise
__global__ void gru_elem(const float* __restrict__ gx, const float* __restrict__ gh,
                         float* __restrict__ h, float* __restrict__ hsum, int total)
{
    int i = blockIdx.x * blockDim.x + threadIdx.x;
    if (i >= total) return;
    int m = i >> 7;
    int c = i & 127;
    const float* gxr = gx + (size_t)m * G3H;
    const float* ghr = gh + (size_t)m * G3H;
    float xr = gxr[c],       xz = gxr[128 + c], xn = gxr[256 + c];
    float hr = ghr[c],       hz = ghr[128 + c], hn = ghr[256 + c];
    float r  = 1.f / (1.f + __expf(-(xr + hr)));
    float z  = 1.f / (1.f + __expf(-(xz + hz)));
    float cn = tanhf(xn + r * hn);
    float hp = h[i];
    float hnew = (1.f - z) * cn + z * hp;
    h[i] = hnew;
    hsum[i] += hnew;
}

// ---------------------------------------------------------------- conv1 edge kernels
__global__ void edge_alpha1(const int* __restrict__ src, const int* __restrict__ dst,
                            const float* __restrict__ eattr,
                            const float* __restrict__ q, const float* __restrict__ k,
                            const float* __restrict__ we,
                            float* __restrict__ alpha, float* __restrict__ mmax, int total)
{
    int t = blockIdx.x * blockDim.x + threadIdx.x;
    if (t >= total) return;
    int e = t >> 2, hd = t & 3;
    int s = src[e], d = dst[e];
    float ea = eattr[e];
    const float* qp = q + (size_t)d * HTOT + hd * 32;
    const float* kp = k + (size_t)s * HTOT + hd * 32;
    const float* wp = we + hd * 32;
    float acc = 0.f;
    #pragma unroll
    for (int i = 0; i < 32; ++i) acc += qp[i] * (kp[i] + ea * wp[i]);
    acc *= 0.17677669529663687f;   // 1/sqrt(32)
    alpha[t] = acc;
    atomicMaxF(&mmax[d * 4 + hd], acc);
}

__global__ void edge_exp1(const int* __restrict__ dst, float* __restrict__ alpha,
                          const float* __restrict__ mmax, float* __restrict__ denom, int total)
{
    int t = blockIdx.x * blockDim.x + threadIdx.x;
    if (t >= total) return;
    int e = t >> 2, hd = t & 3;
    int d = dst[e];
    float a = __expf(alpha[t] - mmax[d * 4 + hd]);
    alpha[t] = a;
    atomicAdd(&denom[d * 4 + hd], a);
}

__global__ void edge_msg1(const int* __restrict__ src, const int* __restrict__ dst,
                          const float* __restrict__ eattr, const float* __restrict__ v,
                          const float* __restrict__ we, const float* __restrict__ alpha,
                          const float* __restrict__ denom, float* __restrict__ agg, int total)
{
    int t = blockIdx.x * blockDim.x + threadIdx.x;
    if (t >= total) return;
    int e = t >> 2, hd = t & 3;
    int s = src[e], d = dst[e];
    float ea = eattr[e];
    float w  = alpha[t] / denom[d * 4 + hd];
    const float* vp = v + (size_t)s * HTOT + hd * 32;
    const float* wp = we + hd * 32;
    float* ap = agg + (size_t)d * HTOT + hd * 32;
    #pragma unroll
    for (int i = 0; i < 32; ++i) atomicAdd(&ap[i], (vp[i] + ea * wp[i]) * w);
}

__global__ void combine1(const float* __restrict__ agg, const float* __restrict__ skip,
                         float* __restrict__ h1, int total)
{
    int i = blockIdx.x * blockDim.x + threadIdx.x;
    if (i >= total) return;
    float x = agg[i] + skip[i];
    h1[i] = (x >= 0.f) ? x : NEG_SLOPE * x;
}

// ---------------------------------------------------------------- output conv
__global__ void out_proj(const float* __restrict__ h1,
                         const float* __restrict__ wq, const float* __restrict__ bq,
                         const float* __restrict__ wk, const float* __restrict__ bk,
                         const float* __restrict__ wv, const float* __restrict__ bv,
                         const float* __restrict__ ws, const float* __restrict__ bs,
                         float* __restrict__ qo, float* __restrict__ ko,
                         float* __restrict__ vo, float* __restrict__ so, int n)
{
    int m = blockIdx.x * blockDim.x + threadIdx.x;
    if (m >= n) return;
    const float* x = h1 + (size_t)m * HTOT;
    float aq = bq[0], ak = bk[0], av = bv[0], as = bs[0];
    #pragma unroll 8
    for (int i = 0; i < HTOT; ++i) {
        float xv = x[i];
        aq += xv * wq[i]; ak += xv * wk[i]; av += xv * wv[i]; as += xv * ws[i];
    }
    qo[m] = aq; ko[m] = ak; vo[m] = av; so[m] = as;
}

__global__ void edge_alpha2(const int* __restrict__ src, const int* __restrict__ dst,
                            const float* __restrict__ eattr, const float* __restrict__ qo,
                            const float* __restrict__ ko, const float* __restrict__ we,
                            float* __restrict__ alpha, float* __restrict__ mo, int total)
{
    int e = blockIdx.x * blockDim.x + threadIdx.x;
    if (e >= total) return;
    int s = src[e], d = dst[e];
    float a = qo[d] * (ko[s] + eattr[e] * we[0]);   // /sqrt(1)
    alpha[e] = a;
    atomicMaxF(&mo[d], a);
}

__global__ void edge_exp2(const int* __restrict__ dst, float* __restrict__ alpha,
                          const float* __restrict__ mo, float* __restrict__ deno, int total)
{
    int e = blockIdx.x * blockDim.x + threadIdx.x;
    if (e >= total) return;
    int d = dst[e];
    float a = __expf(alpha[e] - mo[d]);
    alpha[e] = a;
    atomicAdd(&deno[d], a);
}

__global__ void edge_msg2(const int* __restrict__ src, const int* __restrict__ dst,
                          const float* __restrict__ eattr, const float* __restrict__ vo,
                          const float* __restrict__ we, const float* __restrict__ alpha,
                          const float* __restrict__ deno, float* __restrict__ aggo, int total)
{
    int e = blockIdx.x * blockDim.x + threadIdx.x;
    if (e >= total) return;
    int s = src[e], d = dst[e];
    float w = alpha[e] / deno[d];
    atomicAdd(&aggo[d], (vo[s] + eattr[e] * we[0]) * w);
}

__global__ void final_out(const float* __restrict__ aggo, const float* __restrict__ so,
                          float* __restrict__ out, int n)
{
    int m = blockIdx.x * blockDim.x + threadIdx.x;
    if (m < n) out[m] = aggo[m] + so[m];
}

// ---------------------------------------------------------------- host launcher
extern "C" void kernel_launch(void* const* d_in, const int* in_sizes, int n_in,
                              void* d_out, int out_size, void* d_ws, size_t ws_size,
                              hipStream_t stream) {
    (void)in_sizes; (void)n_in; (void)out_size; (void)ws_size;
    const float* x_seq = (const float*)d_in[0];
    const int*   eidx  = (const int*)d_in[1];
    const int*   src   = eidx;
    const int*   dst   = eidx + NEDGE;
    const float* eattr = (const float*)d_in[2];
    const float* wih = (const float*)d_in[3];
    const float* whh = (const float*)d_in[4];
    const float* bih = (const float*)d_in[5];
    const float* bhh = (const float*)d_in[6];
    const float* c1_wq = (const float*)d_in[7];  const float* c1_bq = (const float*)d_in[8];
    const float* c1_wk = (const float*)d_in[9];  const float* c1_bk = (const float*)d_in[10];
    const float* c1_wv = (const float*)d_in[11]; const float* c1_bv = (const float*)d_in[12];
    const float* c1_we = (const float*)d_in[13];
    const float* c1_ws = (const float*)d_in[14]; const float* c1_bs = (const float*)d_in[15];
    const float* co_wq = (const float*)d_in[16]; const float* co_bq = (const float*)d_in[17];
    const float* co_wk = (const float*)d_in[18]; const float* co_bk = (const float*)d_in[19];
    const float* co_wv = (const float*)d_in[20]; const float* co_bv = (const float*)d_in[21];
    const float* co_we = (const float*)d_in[22];
    const float* co_ws = (const float*)d_in[23]; const float* co_bs = (const float*)d_in[24];
    float* out = (float*)d_out;

    // ---- workspace arena (floats) with phase reuse
    const size_t NH  = (size_t)NNODES * HTOT;     // 2.56M
    const size_t N3H = (size_t)NNODES * G3H;      // 7.68M
    float* W      = (float*)d_ws;
    float* gx     = W;                 // [N,384]  phase2: q | k | v   (3*NH == N3H)
    float* gh     = gx + N3H;          // [N,384]  phase2: skip | agg | h1
    float* h      = gh + N3H;          // [N,128]  later: hmean
    float* hsum   = h + NH;            // [N,128]  later: alphao (E fits)
    float* alpha  = hsum + NH;         // [E,4]
    float* mmax   = alpha + (size_t)NEDGE * 4;    // [N,4]
    float* denom  = mmax + (size_t)NNODES * 4;    // [N,4]
    float* qo     = denom + (size_t)NNODES * 4;   // [N] x7
    float* ko     = qo + NNODES;
    float* vo     = ko + NNODES;
    float* so     = vo + NNODES;
    float* aggo   = so + NNODES;
    float* mo     = aggo + NNODES;
    float* deno   = mo + NNODES;
    // phase2 aliases
    float* q    = gx;            float* k    = gx + NH;   float* v  = gx + 2 * NH;
    float* skip = gh;            float* agg  = gh + NH;   float* h1 = gh + 2 * NH;
    float* alphao = hsum;        // reuse after the time-mean moved into h

    const dim3 blk(256);
    const dim3 gemm384((NNODES + 127) / 128, G3H / 64);   // block tile 128x64
    const dim3 gemm128((NNODES + 127) / 128, HTOT / 64);
    const int totNH = (int)NH;
    const int totE4 = NEDGE * 4;
    auto cdiv = [](int a, int b) { return (a + b - 1) / b; };

    // ---- Phase 1: GRU (h0 = 0), accumulate sum for the time-mean
    fill_kernel<<<cdiv(totNH, 256), blk, 0, stream>>>(h, 0.f, totNH);
    fill_kernel<<<cdiv(totNH, 256), blk, 0, stream>>>(hsum, 0.f, totNH);
    for (int t = 0; t < T_STEPS; ++t) {
        gemm_wmma<DIN, true><<<gemm384, blk, 0, stream>>>(
            x_seq + (size_t)t * NNODES * DIN, wih, bih, gx, NNODES, G3H);
        gemm_wmma<HTOT, true><<<gemm384, blk, 0, stream>>>(
            h, whh, bhh, gh, NNODES, G3H);
        gru_elem<<<cdiv(totNH, 256), blk, 0, stream>>>(gx, gh, h, hsum, totNH);
    }
    scale_kernel<<<cdiv(totNH, 256), blk, 0, stream>>>(h, hsum, 1.f / T_STEPS, totNH);

    // ---- Phase 2: TransformerConv layer 1 (4 heads x 32)
    gemm_wmma<HTOT, false><<<gemm128, blk, 0, stream>>>(h, c1_wq, c1_bq, q,    NNODES, HTOT);
    gemm_wmma<HTOT, false><<<gemm128, blk, 0, stream>>>(h, c1_wk, c1_bk, k,    NNODES, HTOT);
    gemm_wmma<HTOT, false><<<gemm128, blk, 0, stream>>>(h, c1_wv, c1_bv, v,    NNODES, HTOT);
    gemm_wmma<HTOT, false><<<gemm128, blk, 0, stream>>>(h, c1_ws, c1_bs, skip, NNODES, HTOT);

    fill_kernel<<<cdiv(NNODES * 4, 256), blk, 0, stream>>>(mmax, -3.0e38f, NNODES * 4);
    fill_kernel<<<cdiv(NNODES * 4, 256), blk, 0, stream>>>(denom, 0.f, NNODES * 4);
    fill_kernel<<<cdiv(totNH, 256), blk, 0, stream>>>(agg, 0.f, totNH);

    edge_alpha1<<<cdiv(totE4, 256), blk, 0, stream>>>(src, dst, eattr, q, k, c1_we,
                                                      alpha, mmax, totE4);
    edge_exp1<<<cdiv(totE4, 256), blk, 0, stream>>>(dst, alpha, mmax, denom, totE4);
    edge_msg1<<<cdiv(totE4, 256), blk, 0, stream>>>(src, dst, eattr, v, c1_we,
                                                    alpha, denom, agg, totE4);
    combine1<<<cdiv(totNH, 256), blk, 0, stream>>>(agg, skip, h1, totNH);

    // ---- Phase 3: output TransformerConv (1 head, 1 channel)
    out_proj<<<cdiv(NNODES, 256), blk, 0, stream>>>(h1, co_wq, co_bq, co_wk, co_bk,
                                                    co_wv, co_bv, co_ws, co_bs,
                                                    qo, ko, vo, so, NNODES);
    fill_kernel<<<cdiv(NNODES, 256), blk, 0, stream>>>(mo, -3.0e38f, NNODES);
    fill_kernel<<<cdiv(NNODES, 256), blk, 0, stream>>>(deno, 0.f, NNODES);
    fill_kernel<<<cdiv(NNODES, 256), blk, 0, stream>>>(aggo, 0.f, NNODES);

    edge_alpha2<<<cdiv(NEDGE, 256), blk, 0, stream>>>(src, dst, eattr, qo, ko, co_we,
                                                      alphao, mo, NEDGE);
    edge_exp2<<<cdiv(NEDGE, 256), blk, 0, stream>>>(dst, alphao, mo, deno, NEDGE);
    edge_msg2<<<cdiv(NEDGE, 256), blk, 0, stream>>>(src, dst, eattr, vo, co_we,
                                                    alphao, deno, aggo, NEDGE);
    final_out<<<cdiv(NNODES, 256), blk, 0, stream>>>(aggo, so, out, NNODES);
}
